// MultiHopReasoningGNN_21689584845618
// MI455X (gfx1250) — matrix operand order repaired and checked by reading, not compiled, not run on hardware
//
#include <hip/hip_runtime.h>
#include <hip/hip_bf16.h>

#define N_NODES 50000
#define N_EDGES 400000
#define IN_DIM  128
#define HID     64
#define HEADS   4
#define HC      256     // HEADS * 64
#define EDIM    16
#define LAYERS  6
#define NEG_SLOPE 0.2f
#define MTILES  (N_NODES / 16)   // 3125

typedef __attribute__((ext_vector_type(16))) _Float16 v16h;
typedef __attribute__((ext_vector_type(8)))  _Float16 v8h;
typedef __attribute__((ext_vector_type(4)))  _Float16 v4h;
typedef __attribute__((ext_vector_type(8)))  float    v8f;

// Build A fragment (16x32 f16): lane holds row M=lane%16,
// element e -> K = kbase + e, kbase includes
// the (lane<16 ? 0 : 16) half-wave offset. Two adjacent 16B LDS chunks.
__device__ __forceinline__ v16h load_a_frag(const _Float16* __restrict__ row, int kbase) {
  v8h lo = *(const v8h*)(row + kbase);
  v8h hi = *(const v8h*)(row + kbase + 8);
  return __builtin_shufflevector(lo, hi, 0,1,2,3,4,5,6,7,8,9,10,11,12,13,14,15);
}

__device__ __forceinline__ v4h cvt4(float4 v) {
  v4h r = {(_Float16)v.x, (_Float16)v.y, (_Float16)v.z, (_Float16)v.w};
  return r;
}

// ---------------------------------------------------------------------------
// Input projection: h0 = x @ in_W^T + in_b   [50000,128] x [128,64]
// 256 threads = 8 waves; wave w owns M-tile (blockIdx.x*8+w) and sweeps all
// 4 N-tiles against block-shared LDS weights (8x weight-traffic amortization).
// ---------------------------------------------------------------------------
__global__ void __launch_bounds__(256)
k_in_proj(const float* __restrict__ x, const float* __restrict__ W,
          const float* __restrict__ bias, float* __restrict__ out)
{
  __shared__ __align__(32) _Float16 Ws[HID][IN_DIM];    // 16 KB
  __shared__ __align__(32) _Float16 As[8][16][IN_DIM];  // 32 KB
  const int tid   = threadIdx.x;
  const int lane  = tid & 31;
  const int wave  = tid >> 5;
  const int mtile = blockIdx.x * 8 + wave;

  for (int i = tid; i < HID * IN_DIM / 4; i += 256) {
    int r = i >> 5, k = (i & 31) * 4;
    *(v4h*)&Ws[r][k] = cvt4(*(const float4*)&W[r * IN_DIM + k]);
  }
  if (mtile < MTILES) {
    for (int i = lane; i < 16 * IN_DIM / 4; i += 32) {
      int r = i >> 5, k = (i & 31) * 4;
      *(v4h*)&As[wave][r][k] =
          cvt4(*(const float4*)&x[(size_t)(mtile * 16 + r) * IN_DIM + k]);
    }
  }
  __syncthreads();
  if (mtile >= MTILES) return;

  const int lrow  = lane & 15;
  const int lhalf = lane >> 4;

  v16h a[4];
  #pragma unroll
  for (int ks = 0; ks < 4; ++ks)
    a[ks] = load_a_frag(&As[wave][lrow][0], ks * 32 + lhalf * 16);

  #pragma unroll
  for (int nt = 0; nt < 4; ++nt) {
    const _Float16* wrow = &Ws[nt * 16 + lrow][0];
    v8f acc = {};
    #pragma unroll
    for (int ks = 0; ks < 4; ++ks) {
      v16h b = *(const v16h*)(wrow + ks * 32 + lhalf * 16);
      acc = __builtin_amdgcn_wmma_f32_16x16x32_f16(false, a[ks], false, b,
                                                   (short)0, acc, false, false);
    }
    const int col = nt * 16 + lrow;
    const float bc = bias[col];
    #pragma unroll
    for (int r = 0; r < 8; ++r) {
      int row = mtile * 16 + r + 8 * lhalf;
      out[(size_t)row * HID + col] = acc[r] + bc;
    }
  }
}

// ---------------------------------------------------------------------------
// Layer projections: xl = h @ Wl^T + bl, xr = h @ Wr^T + br (blockIdx.y sel)
// [50000,64] x [64,256]; 8 waves/block, wave per M-tile, 16 N-tiles each.
// ---------------------------------------------------------------------------
__global__ void __launch_bounds__(256)
k_layer_proj(const float* __restrict__ h,
             const float* __restrict__ Wl, const float* __restrict__ bl,
             const float* __restrict__ Wr, const float* __restrict__ br,
             float* __restrict__ xl, float* __restrict__ xr)
{
  __shared__ __align__(32) _Float16 Ws[HC][HID];     // 32 KB
  __shared__ __align__(32) _Float16 As[8][16][HID];  // 16 KB
  const float* W   = blockIdx.y ? Wr : Wl;
  const float* bb  = blockIdx.y ? br : bl;
  float*       out = blockIdx.y ? xr : xl;

  const int tid   = threadIdx.x;
  const int lane  = tid & 31;
  const int wave  = tid >> 5;
  const int mtile = blockIdx.x * 8 + wave;

  for (int i = tid; i < HC * HID / 4; i += 256) {
    int r = i >> 4, k = (i & 15) * 4;
    *(v4h*)&Ws[r][k] = cvt4(*(const float4*)&W[r * HID + k]);
  }
  if (mtile < MTILES) {
    for (int i = lane; i < 16 * HID / 4; i += 32) {
      int r = i >> 4, k = (i & 15) * 4;
      *(v4h*)&As[wave][r][k] =
          cvt4(*(const float4*)&h[(size_t)(mtile * 16 + r) * HID + k]);
    }
  }
  __syncthreads();
  if (mtile >= MTILES) return;

  const int lrow  = lane & 15;
  const int lhalf = lane >> 4;

  const v16h a0 = load_a_frag(&As[wave][lrow][0],  0 + lhalf * 16);
  const v16h a1 = load_a_frag(&As[wave][lrow][0], 32 + lhalf * 16);

  #pragma unroll
  for (int nt = 0; nt < HC / 16; ++nt) {
    const _Float16* wrow = &Ws[nt * 16 + lrow][0];
    v8f acc = {};
    v16h b0 = *(const v16h*)(wrow +  0 + lhalf * 16);
    acc = __builtin_amdgcn_wmma_f32_16x16x32_f16(false, a0, false, b0,
                                                 (short)0, acc, false, false);
    v16h b1 = *(const v16h*)(wrow + 32 + lhalf * 16);
    acc = __builtin_amdgcn_wmma_f32_16x16x32_f16(false, a1, false, b1,
                                                 (short)0, acc, false, false);
    const int col = nt * 16 + lrow;
    const float bc = bb[col];
    #pragma unroll
    for (int r = 0; r < 8; ++r) {
      int row = mtile * 16 + r + 8 * lhalf;
      out[(size_t)row * HC + col] = acc[r] + bc;
    }
  }
}

// etp[3][256] = edge_table[3][16] @ We^T   (tiny, one block)
__global__ void k_etp(const float* __restrict__ etab, const float* __restrict__ We,
                      float* __restrict__ etp)
{
  const int t = threadIdx.x;  // 0..255 output channel
  #pragma unroll
  for (int r = 0; r < 3; ++r) {
    float acc = 0.f;
    #pragma unroll
    for (int k = 0; k < EDIM; ++k) acc += etab[r * EDIM + k] * We[t * EDIM + k];
    etp[r * HC + t] = acc;
  }
}

// ---------------------------------------------------------------------------
// Edge pass 1: logits -> exp -> den atomic (softmax shift skipped: logits
// are O(0.1), softmax is shift-invariant). One wave32 per edge; lane handles
// 8 contiguous channels; per-head (8-lane) reduction via shfl_xor.
// ---------------------------------------------------------------------------
__global__ void __launch_bounds__(256)
k_edge_logits(const int* __restrict__ ei, const int* __restrict__ eattr,
              const float* __restrict__ xl, const float* __restrict__ xr,
              const float* __restrict__ etp, const float* __restrict__ att,
              float* __restrict__ exb, float* __restrict__ den)
{
  const int e = blockIdx.x * 8 + (threadIdx.x >> 5);
  const int lane = threadIdx.x & 31;
  const int src = ei[e];
  const int dst = ei[N_EDGES + e];
  const int at  = eattr[e];
  const int c0   = lane * 8;
  const int head = lane >> 3;

  const float4* pl = (const float4*)(xl + (size_t)src * HC + c0);
  const float4* pr = (const float4*)(xr + (size_t)dst * HC + c0);
  const float4* pe = (const float4*)(etp + at * HC + c0);
  const float4* pa = (const float4*)(att + head * 64 + (c0 & 63));

  float lacc = 0.f;
  #pragma unroll
  for (int j = 0; j < 2; ++j) {
    float4 a = pl[j], b = pr[j], t = pe[j], w = pa[j];
    float m;
    m = a.x + b.x + t.x; lacc += (m > 0.f ? m : NEG_SLOPE * m) * w.x;
    m = a.y + b.y + t.y; lacc += (m > 0.f ? m : NEG_SLOPE * m) * w.y;
    m = a.z + b.z + t.z; lacc += (m > 0.f ? m : NEG_SLOPE * m) * w.z;
    m = a.w + b.w + t.w; lacc += (m > 0.f ? m : NEG_SLOPE * m) * w.w;
  }
  lacc += __shfl_xor(lacc, 1, 32);
  lacc += __shfl_xor(lacc, 2, 32);
  lacc += __shfl_xor(lacc, 4, 32);
  if ((lane & 7) == 0) {
    const float ex = __expf(lacc);
    exb[(size_t)e * HEADS + head] = ex;
    atomicAdd(&den[(size_t)dst * HEADS + head], ex);
  }
}

// Edge pass 2: agg[dst] += alpha * xl[src]   (L2-resident scatter)
__global__ void __launch_bounds__(256)
k_aggregate(const int* __restrict__ ei, const float* __restrict__ xl,
            const float* __restrict__ exb, const float* __restrict__ den,
            float* __restrict__ agg)
{
  const int e = blockIdx.x * 8 + (threadIdx.x >> 5);
  const int lane = threadIdx.x & 31;
  const int src = ei[e];
  const int dst = ei[N_EDGES + e];
  const int head = lane >> 3;
  const float alpha = exb[(size_t)e * HEADS + head] /
                      fmaxf(den[(size_t)dst * HEADS + head], 1e-16f);
  const int c0 = lane * 8;
  const float4* pl = (const float4*)(xl + (size_t)src * HC + c0);
  float* pd = agg + (size_t)dst * HC + c0;
  #pragma unroll
  for (int j = 0; j < 2; ++j) {
    float4 v = pl[j];
    atomicAdd(pd + j * 4 + 0, v.x * alpha);
    atomicAdd(pd + j * 4 + 1, v.y * alpha);
    atomicAdd(pd + j * 4 + 2, v.z * alpha);
    atomicAdd(pd + j * 4 + 3, v.w * alpha);
  }
}

// Head-mean + bias + LayerNorm (+residual, +relu). One wave per node.
__global__ void __launch_bounds__(256)
k_finish(const float* __restrict__ agg, const float* __restrict__ h_in,
         const float* __restrict__ out_bias, const float* __restrict__ ln_g,
         const float* __restrict__ ln_b, float* __restrict__ h_out,
         int use_res, int use_relu)
{
  const int n = blockIdx.x * 8 + (threadIdx.x >> 5);
  const int lane = threadIdx.x & 31;
  const float* base = agg + (size_t)n * HC;

  float v[2];
  #pragma unroll
  for (int t = 0; t < 2; ++t) {
    const int c = lane + t * 32;
    const float s = base[c] + base[64 + c] + base[128 + c] + base[192 + c];
    v[t] = 0.25f * s + out_bias[c];
  }
  float sum = v[0] + v[1];
  #pragma unroll
  for (int off = 16; off; off >>= 1) sum += __shfl_xor(sum, off, 32);
  const float mu = sum * (1.f / 64.f);
  const float d0 = v[0] - mu, d1 = v[1] - mu;
  float vs = d0 * d0 + d1 * d1;
  #pragma unroll
  for (int off = 16; off; off >>= 1) vs += __shfl_xor(vs, off, 32);
  const float inv = rsqrtf(vs * (1.f / 64.f) + 1e-5f);

  #pragma unroll
  for (int t = 0; t < 2; ++t) {
    const int c = lane + t * 32;
    float o = (v[t] - mu) * inv * ln_g[c] + ln_b[c];
    if (use_res)  o += h_in[(size_t)n * HID + c];
    if (use_relu) o = fmaxf(o, 0.f);
    h_out[(size_t)n * HID + c] = o;
  }
}

extern "C" void kernel_launch(void* const* d_in, const int* in_sizes, int n_in,
                              void* d_out, int out_size, void* d_ws, size_t ws_size,
                              hipStream_t stream)
{
  (void)in_sizes; (void)n_in; (void)out_size; (void)ws_size;
  const float* x        = (const float*)d_in[0];
  const int*   ei       = (const int*)  d_in[1];
  const int*   eattr    = (const int*)  d_in[2];
  const float* in_W     = (const float*)d_in[3];
  const float* in_b     = (const float*)d_in[4];
  const float* etab     = (const float*)d_in[5];
  const float* Wl       = (const float*)d_in[6];
  const float* bl       = (const float*)d_in[7];
  const float* Wr       = (const float*)d_in[8];
  const float* br       = (const float*)d_in[9];
  const float* We       = (const float*)d_in[10];
  const float* att      = (const float*)d_in[11];
  const float* out_bias = (const float*)d_in[12];
  const float* ln_g     = (const float*)d_in[13];
  const float* ln_b     = (const float*)d_in[14];

  // Workspace carving (~122 MB). agg aliases xr: xr is dead after k_edge_logits.
  char* ws = (char*)d_ws;
  size_t off = 0;
  auto carve = [&](size_t bytes) -> char* {
    char* p = ws + off;
    off += (bytes + 255) & ~(size_t)255;
    return p;
  };
  float* xl  = (float*)carve((size_t)N_NODES * HC * sizeof(float));
  float* xr  = (float*)carve((size_t)N_NODES * HC * sizeof(float));
  float* agg = xr;
  float* exb = (float*)carve((size_t)N_EDGES * HEADS * sizeof(float));
  float* den = (float*)carve((size_t)N_NODES * HEADS * sizeof(float));
  float* etp = (float*)carve(3 * HC * sizeof(float));
  float* hws = (float*)carve((size_t)N_NODES * HID * sizeof(float));

  float* hA = (float*)d_out;  // holds h0, h2, h4, h6 (final lands in d_out)
  float* hB = hws;            // holds h1, h3, h5

  const int gemm_blocks = (MTILES + 7) / 8;   // 391

  k_in_proj<<<gemm_blocks, 256, 0, stream>>>(x, in_W, in_b, hA);

  for (int i = 0; i < LAYERS; ++i) {
    const float* hin  = (i & 1) ? hB : hA;
    float*       hout = (i & 1) ? hA : hB;

    dim3 g(gemm_blocks, 2);
    k_layer_proj<<<g, 256, 0, stream>>>(hin,
                                        Wl + (size_t)i * HC * HID, bl + (size_t)i * HC,
                                        Wr + (size_t)i * HC * HID, br + (size_t)i * HC,
                                        xl, xr);
    k_etp<<<1, HC, 0, stream>>>(etab, We + (size_t)i * HC * EDIM, etp);

    hipMemsetAsync(den, 0, (size_t)N_NODES * HEADS * sizeof(float), stream);
    k_edge_logits<<<N_EDGES / 8, 256, 0, stream>>>(ei, eattr, xl, xr, etp,
                                                   att + (size_t)i * HEADS * 64,
                                                   exb, den);
    hipMemsetAsync(agg, 0, (size_t)N_NODES * HC * sizeof(float), stream);
    k_aggregate<<<N_EDGES / 8, 256, 0, stream>>>(ei, xl, exb, den, agg);

    k_finish<<<N_NODES / 8, 256, 0, stream>>>(agg, hin,
                                              out_bias + i * HID,
                                              ln_g + i * HID, ln_b + i * HID,
                                              hout,
                                              (i >= LAYERS / 2) ? 1 : 0,
                                              (i < LAYERS - 1) ? 1 : 0);
  }
}
